// MultiHeadAttention_8924942041462
// MI455X (gfx1250) — matrix operand order repaired and checked by reading, not compiled, not run on hardware
//
#include <hip/hip_runtime.h>

// ---------------- types ----------------
typedef __bf16 bf16_t;
typedef __attribute__((ext_vector_type(16))) __bf16 v16bf;
typedef __attribute__((ext_vector_type(8)))  __bf16 v8bf;
typedef __attribute__((ext_vector_type(4)))  __bf16 v4bf;
typedef __attribute__((ext_vector_type(8)))  float  v8f;

#define B_  2
#define S_  2048
#define E_  1024
#define H_  16
#define D_  64
#define SCALE_ 0.125f   // D^-0.5 = 64^-0.5

__device__ __forceinline__ bf16_t f2b(float f) {
  unsigned u = __builtin_bit_cast(unsigned, f);
  u += 0x7FFFu + ((u >> 16) & 1u);           // round-to-nearest-even
  unsigned short s = (unsigned short)(u >> 16);
  return __builtin_bit_cast(bf16_t, s);
}

__device__ __forceinline__ v8f vzero8() {
  v8f z;
#pragma unroll
  for (int i = 0; i < 8; ++i) z[i] = 0.f;
  return z;
}

// ---------------- fragment loaders (ISA 7.12.2 layouts) ----------------
// A-matrix 16x32 bf16: lane l: row M = l&15; half h = l>>4;
//   v0..v3 hold K = k0 + h*8 + [0..7]   (pairs per VGPR)
//   v4..v7 hold K = k0 + 16 + h*8 + [0..7]
__device__ __forceinline__ v16bf load_a(const bf16_t* base, int ld, int k0, int lane) {
  int r = lane & 15, h = lane >> 4;
  const bf16_t* p = base + (size_t)r * ld + k0 + h * 8;
  v8bf lo = *(const v8bf*)(p);
  v8bf hi = *(const v8bf*)(p + 16);
  return __builtin_shufflevector(lo, hi, 0,1,2,3,4,5,6,7,8,9,10,11,12,13,14,15);
}

// B-matrix 32x16 bf16: lane l: col N = l&15; half h = l>>4;
//   lane holds contraction K = k0 + h*8 + [0..7] (v0..v3), k0 + 16 + h*8 + [0..7] (v4..v7) of column N
// Source "column N" here is a row of a K-contiguous buffer (W row / K row / V^T row).
__device__ __forceinline__ v16bf load_b(const bf16_t* base, int ld, int k0, int lane) {
  int n = lane & 15, h = lane >> 4;
  const bf16_t* p = base + (size_t)n * ld + k0 + h * 8;
  v8bf lo = *(const v8bf*)(p);
  v8bf hi = *(const v8bf*)(p + 16);
  return __builtin_shufflevector(lo, hi, 0,1,2,3,4,5,6,7,8,9,10,11,12,13,14,15);
}

__device__ __forceinline__ v8f wmma_bf(v16bf a, v16bf b, v8f c) {
  return __builtin_amdgcn_wmma_f32_16x16x32_bf16(false, a, false, b, (short)0, c, false, false);
}

// ---------------- fp32 -> bf16 conversion ----------------
__global__ void cvt_bf16(const float* __restrict__ in, bf16_t* __restrict__ out, int n4) {
  int i = blockIdx.x * blockDim.x + threadIdx.x;
  if (i < n4) {
    float4 f = ((const float4*)in)[i];
    v4bf o; o[0] = f2b(f.x); o[1] = f2b(f.y); o[2] = f2b(f.z); o[3] = f2b(f.w);
    ((v4bf*)out)[i] = o;
  }
}

// ---------------- projection GEMM:  out = A[4096,1024] @ W^T + bias ----------------
// mode 0: Q -> bf16 [B,H,S,D]; mode 1: K -> bf16 [B,H,S,D];
// mode 2: V -> bf16 [B,H,D,S] (transposed); mode 3: final -> f32 row-major [4096,1024]
__global__ void __launch_bounds__(128)
proj_gemm(const bf16_t* __restrict__ A, const bf16_t* __restrict__ W,
          const float* __restrict__ bias, bf16_t* __restrict__ outb,
          float* __restrict__ outf, int mode)
{
  int lane = threadIdx.x & 31;
  int w    = threadIdx.x >> 5;                 // 4 waves, 2x2 arrangement
  int mbase = blockIdx.x * 128 + (w & 1) * 64; // wave tile: 64x64
  int nbase = blockIdx.y * 128 + (w >> 1) * 64;

  v8f c[4][4];
#pragma unroll
  for (int i = 0; i < 4; ++i)
#pragma unroll
    for (int j = 0; j < 4; ++j) c[i][j] = vzero8();

  for (int k0 = 0; k0 < E_; k0 += 32) {
    v16bf a[4], b[4];
#pragma unroll
    for (int i = 0; i < 4; ++i) a[i] = load_a(A + (size_t)(mbase + i * 16) * E_, E_, k0, lane);
#pragma unroll
    for (int j = 0; j < 4; ++j) b[j] = load_b(W + (size_t)(nbase + j * 16) * E_, E_, k0, lane);
#pragma unroll
    for (int i = 0; i < 4; ++i)
#pragma unroll
      for (int j = 0; j < 4; ++j) c[i][j] = wmma_bf(a[i], b[j], c[i][j]);
  }

  // C/D layout: lane<16: N=lane, M=vgpr; lane>=16: N=lane-16, M=vgpr+8
  int nn = lane & 15, hh = lane >> 4;
#pragma unroll
  for (int i = 0; i < 4; ++i)
#pragma unroll
    for (int j = 0; j < 4; ++j) {
      int col = nbase + j * 16 + nn;
      float bv = bias[col];
#pragma unroll
      for (int e = 0; e < 8; ++e) {
        int row = mbase + i * 16 + e + 8 * hh;       // token index
        float val = c[i][j][e] + bv;
        if (mode == 3) {
          outf[(size_t)row * E_ + col] = val;
        } else {
          int bi = row >> 11, s = row & (S_ - 1);    // S_=2048
          int hd = col >> 6,  d = col & (D_ - 1);
          if (mode == 2)
            outb[((size_t)(bi * H_ + hd) * D_ + d) * S_ + s] = f2b(val);   // V^T
          else
            outb[((size_t)(bi * H_ + hd) * S_ + s) * D_ + d] = f2b(val);   // Q,K
        }
      }
    }
}

// ---------------- flash attention: one wave per 32-query tile ----------------
// K/V fragments per 32-key chunk are reused by two 16-query sub-tiles:
// inner loop = 16 WMMAs per 8 global fragment loads.
__global__ void __launch_bounds__(128)
flash_attn(const bf16_t* __restrict__ Q,   // [B*H, S, D]
           const bf16_t* __restrict__ K,   // [B*H, S, D]
           const bf16_t* __restrict__ Vt,  // [B*H, D, S]
           bf16_t* __restrict__ Obuf)      // [B*S, E]
{
  __shared__ __align__(32) bf16_t pst[4][2 * 16 * 32];  // per-wave P staging (2 sub-tiles)
  int lane = threadIdx.x & 31;
  int w    = threadIdx.x >> 5;
  int tile = blockIdx.x * 4 + w;           // 0..2047 (32-query tiles)
  int bh    = tile >> 6;                   // 64 query tiles per (b,h)
  int qbase = (tile & 63) * 32;

  const bf16_t* Qb = Q  + (size_t)bh * S_ * D_ + (size_t)qbase * D_;
  const bf16_t* Kb = K  + (size_t)bh * S_ * D_;
  const bf16_t* Vb = Vt + (size_t)bh * D_ * S_;

  v16bf qf[2][2];
#pragma unroll
  for (int qt = 0; qt < 2; ++qt) {
    qf[qt][0] = load_a(Qb + (size_t)qt * 16 * D_, D_, 0,  lane);
    qf[qt][1] = load_a(Qb + (size_t)qt * 16 * D_, D_, 32, lane);
  }

  v8f o[2][4];
#pragma unroll
  for (int qt = 0; qt < 2; ++qt)
#pragma unroll
    for (int j = 0; j < 4; ++j) o[qt][j] = vzero8();
  float m[2][8], lsum[2][8];
#pragma unroll
  for (int qt = 0; qt < 2; ++qt)
#pragma unroll
    for (int e = 0; e < 8; ++e) { m[qt][e] = -3.0e38f; lsum[qt][e] = 0.f; }

  int nn = lane & 15, hh = lane >> 4;
  bf16_t* lp = pst[w];

  for (int kc = 0; kc < S_; kc += 32) {
    // prefetch next K/V chunk (global_prefetch_b8); 32 lanes cover the 4KB chunks
    int pk = (kc + 32 < S_) ? (kc + 32) : kc;
    __builtin_prefetch(Kb + (size_t)(pk + lane) * D_, 0, 3);
    __builtin_prefetch(Vb + (size_t)lane * S_ + pk, 0, 3);
    __builtin_prefetch(Vb + (size_t)(lane + 32) * S_ + pk, 0, 3);

    // ---- shared K/V fragments for this chunk ----
    v16bf kb[2][2];
#pragma unroll
    for (int kt = 0; kt < 2; ++kt)
#pragma unroll
      for (int kh = 0; kh < 2; ++kh)
        kb[kt][kh] = load_b(Kb + (size_t)(kc + kt * 16) * D_, D_, kh * 32, lane);
    v16bf vb[4];
#pragma unroll
    for (int j = 0; j < 4; ++j)
      vb[j] = load_b(Vb + (size_t)(j * 16) * S_ + kc, S_, 0, lane);

    // ---- scores + online softmax per 16-query sub-tile ----
#pragma unroll
    for (int qt = 0; qt < 2; ++qt) {
      v8f s0 = vzero8(), s1 = vzero8();
      s0 = wmma_bf(qf[qt][0], kb[0][0], s0);
      s0 = wmma_bf(qf[qt][1], kb[0][1], s0);
      s1 = wmma_bf(qf[qt][0], kb[1][0], s1);
      s1 = wmma_bf(qf[qt][1], kb[1][1], s1);

      float alpha[8];
#pragma unroll
      for (int e = 0; e < 8; ++e) {
        float v0 = s0[e] * SCALE_, v1 = s1[e] * SCALE_;
        float mx = fmaxf(v0, v1);
#pragma unroll
        for (int off = 1; off < 16; off <<= 1) mx = fmaxf(mx, __shfl_xor(mx, off, 32));
        float mn = fmaxf(m[qt][e], mx);
        float a  = __expf(m[qt][e] - mn);
        m[qt][e] = mn;
        float p0 = __expf(v0 - mn), p1 = __expf(v1 - mn);
        float rs = p0 + p1;
#pragma unroll
        for (int off = 1; off < 16; off <<= 1) rs += __shfl_xor(rs, off, 32);
        lsum[qt][e] = lsum[qt][e] * a + rs;
        alpha[e] = a;
        s0[e] = p0; s1[e] = p1;
      }
#pragma unroll
      for (int j = 0; j < 4; ++j)
#pragma unroll
        for (int e = 0; e < 8; ++e) o[qt][j][e] *= alpha[e];

      // C-layout P (16x32) -> LDS row-major (sub-tile qt)
      bf16_t* lq = lp + qt * (16 * 32);
#pragma unroll
      for (int e = 0; e < 8; ++e) {
        int row = e + 8 * hh;
        lq[row * 32 + nn]      = f2b(s0[e]);
        lq[row * 32 + 16 + nn] = f2b(s1[e]);
      }
    }

    // ---- O += P(16x32) @ V_chunk(32x64) for both sub-tiles (vb reused) ----
#pragma unroll
    for (int qt = 0; qt < 2; ++qt) {
      v16bf pf = load_a(lp + qt * (16 * 32), 32, 0, lane);  // DScnt wait by compiler
#pragma unroll
      for (int j = 0; j < 4; ++j) o[qt][j] = wmma_bf(pf, vb[j], o[qt][j]);
    }
  }

  // ---- finalize: divide by row sum, scatter to [B*S, E] ----
  int bi = bh >> 4, hd = bh & (H_ - 1);
#pragma unroll
  for (int qt = 0; qt < 2; ++qt)
#pragma unroll
    for (int j = 0; j < 4; ++j)
#pragma unroll
      for (int e = 0; e < 8; ++e) {
        int s = qbase + qt * 16 + e + 8 * hh;
        int d = j * 16 + nn;
        float val = o[qt][j][e] / lsum[qt][e];
        Obuf[((size_t)(bi * S_ + s)) * E_ + hd * D_ + d] = f2b(val);
      }
}

// ---------------- host ----------------
extern "C" void kernel_launch(void* const* d_in, const int* in_sizes, int n_in,
                              void* d_out, int out_size, void* d_ws, size_t ws_size,
                              hipStream_t stream) {
  (void)in_sizes; (void)n_in; (void)out_size; (void)ws_size;
  const float* x  = (const float*)d_in[0];
  const float* Wq = (const float*)d_in[1];
  const float* bq = (const float*)d_in[2];
  const float* Wk = (const float*)d_in[3];
  const float* bk = (const float*)d_in[4];
  const float* Wv = (const float*)d_in[5];
  const float* bv = (const float*)d_in[6];
  const float* Wo = (const float*)d_in[7];
  const float* bo = (const float*)d_in[8];
  float* out = (float*)d_out;

  size_t off = 0;
  auto carve = [&](size_t bytes) -> void* {
    void* p = (char*)d_ws + off;
    off += (bytes + 255) & ~(size_t)255;
    return p;
  };
  const size_t NT = (size_t)B_ * S_;                 // 4096 tokens
  bf16_t* Xb   = (bf16_t*)carve(NT * E_ * 2);
  bf16_t* Wqb  = (bf16_t*)carve((size_t)E_ * E_ * 2);
  bf16_t* Wkb  = (bf16_t*)carve((size_t)E_ * E_ * 2);
  bf16_t* Wvb  = (bf16_t*)carve((size_t)E_ * E_ * 2);
  bf16_t* Wob  = (bf16_t*)carve((size_t)E_ * E_ * 2);
  bf16_t* Qbuf = (bf16_t*)carve(NT * E_ * 2);        // [B,H,S,D]
  bf16_t* Kbuf = (bf16_t*)carve(NT * E_ * 2);        // [B,H,S,D]
  bf16_t* Vtb  = (bf16_t*)carve(NT * E_ * 2);        // [B,H,D,S]
  bf16_t* Attn = (bf16_t*)carve(NT * E_ * 2);        // [B*S, E]

  // fp32 -> bf16
  {
    int n4x = (int)(NT * E_ / 4);                    // 1,048,576
    int n4w = E_ * E_ / 4;                           // 262,144
    cvt_bf16<<<(n4x + 255) / 256, 256, 0, stream>>>(x,  Xb,  n4x);
    cvt_bf16<<<(n4w + 255) / 256, 256, 0, stream>>>(Wq, Wqb, n4w);
    cvt_bf16<<<(n4w + 255) / 256, 256, 0, stream>>>(Wk, Wkb, n4w);
    cvt_bf16<<<(n4w + 255) / 256, 256, 0, stream>>>(Wv, Wvb, n4w);
    cvt_bf16<<<(n4w + 255) / 256, 256, 0, stream>>>(Wo, Wob, n4w);
  }

  dim3 gGemm(NT / 128, E_ / 128);                    // 32 x 8 blocks, 128 thr (4 waves)
  proj_gemm<<<gGemm, 128, 0, stream>>>(Xb, Wqb, bq, Qbuf, nullptr, 0);
  proj_gemm<<<gGemm, 128, 0, stream>>>(Xb, Wkb, bk, Kbuf, nullptr, 1);
  proj_gemm<<<gGemm, 128, 0, stream>>>(Xb, Wvb, bv, Vtb,  nullptr, 2);

  flash_attn<<<(B_ * H_ * (S_ / 32)) / 4, 128, 0, stream>>>(Qbuf, Kbuf, Vtb, Attn);

  proj_gemm<<<gGemm, 128, 0, stream>>>(Attn, Wob, bo, nullptr, out, 3);
}
